// Decay_60155311947799
// MI455X (gfx1250) — matrix operand order repaired and checked by reading, not compiled, not run on hardware
//
#include <hip/hip_runtime.h>
#include <hip/hip_bf16.h>
#include <stdint.h>

typedef __attribute__((ext_vector_type(16))) _Float16 v16h;
typedef __attribute__((ext_vector_type(8)))  float    v8f;

#define Bn 2
#define Pn 4
#define Cn 3
#define Hn 256
#define Wn 256
#define HW (Hn*Wn)
#define Tn 27           // 27 tao values, padded to K=32 for WMMA
#define BGHn 2764
#define BGWn 3856

#if defined(__AMDGCN__) && __has_builtin(__builtin_amdgcn_global_load_async_to_lds_b32) && __has_builtin(__builtin_amdgcn_s_wait_asynccnt)
#define USE_ASYNC_LDS 1
typedef __attribute__((address_space(1))) int as1_int;
typedef __attribute__((address_space(3))) int as3_int;
#else
#define USE_ASYNC_LDS 0
#endif

// ---------- stateless counter-based RNG (graph-capture safe) ----------
__device__ __forceinline__ uint32_t hash32(uint64_t v) {
  v ^= v >> 33; v *= 0xff51afd7ed558ccdULL;
  v ^= v >> 33; v *= 0xc4ceb9fe1a85ec53ULL;
  v ^= v >> 33;
  return (uint32_t)v;
}
__device__ __forceinline__ float u01(uint64_t v) {
  return (float)(hash32(v) >> 8) * (1.0f/16777216.0f);
}
__device__ __forceinline__ float poisson_knuth(float lam, uint64_t seed) {
  const float L = __expf(-lam);
  float prod = 1.0f;
  float k = 0.0f;
  #pragma unroll 1
  for (int it = 0; it < 16; ++it) {
    prod *= u01(seed + (uint64_t)it * 0x9E3779B9ULL);
    if (prod <= L) break;
    k += 1.0f;
  }
  return k;
}

// ---------------------------------------------------------------------
// Pass 1: fused w_i + normalization + decay einsum + tao_gt + x2 + block max.
// Per wave: 16 pixels. A-matrix (16x32 f16) = tao weights; B col0 = et[k,p],
// B col1 = valid-ones. One v_wmma_f32_16x16x32_f16 gives both the decay
// numerator and the normalization sum S for 16 pixels at once, on the matrix
// pipe, co-executing with the v_exp_f32 TRANS stream.
// x is staged block-wide into LDS via GLOBAL_LOAD_ASYNC_TO_LDS (ASYNCcnt).
// ---------------------------------------------------------------------
__global__ __launch_bounds__(256)
void decay_pass1(const float* __restrict__ x, const float* __restrict__ t,
                 const float* __restrict__ ratio, const float* __restrict__ life,
                 const float* __restrict__ inten,
                 float* __restrict__ tao_gt, float* __restrict__ x2buf,
                 float* __restrict__ blockMax)
{
  __shared__ float xTile[Pn*Cn][128];   // 12 slices x 128 pixels = 6 KB
  __shared__ float ldsNum[8][16];
  __shared__ float ldsS[8][16];
  __shared__ float ldsWMax[8];

  const int lane = threadIdx.x & 31;
  const int wave = threadIdx.x >> 5;
  const int m    = lane & 15;   // pixel within 16-wide tile (WMMA row M)
  const int kg   = lane >> 4;   // which half of K this lane owns
  const int b    = blockIdx.y;
  const int blockBase = blockIdx.x * 128;
  const int pixLocal  = wave * 16 + m;            // 0..127
  const int pix       = blockBase + pixLocal;     // 0..65535

  // ---- stage x[b, :, :, blockBase .. blockBase+127] into LDS ----
  {
    const int tid = threadIdx.x;
    #pragma unroll
    for (int it = 0; it < 6; ++it) {
      const int lin   = it * 256 + tid;   // 0..1535
      const int slice = lin >> 7;         // p*3+c
      const int elem  = lin & 127;
      const float* g = x + (((size_t)b*(Pn*Cn) + slice) << 16) + blockBase + elem;
#if USE_ASYNC_LDS
      __builtin_amdgcn_global_load_async_to_lds_b32(
          (as1_int*)g, (as3_int*)&xTile[slice][elem], 0, 0);
#else
      xTile[slice][elem] = *g;
#endif
    }
#if USE_ASYNC_LDS
    __builtin_amdgcn_s_wait_asynccnt(0);
#endif
  }
  __syncthreads();

  const float life0 = life[0], life1 = life[1], life2 = life[2];
  const float int0 = inten[0], int1 = inten[1], int2 = inten[2];
  const float coef = rsqrtf(2.0f * 3.14f);   // reference uses 3.14

  // A-matrix K ownership (ISA 7.12.2, 16-bit A 16x32):
  // lanes 0-15: halves -> K {0..7, 16..23}; lanes 16-31: K {8..15, 24..31}
  int   myK[16];
  float myTao[16];
  #pragma unroll
  for (int hi = 0; hi < 16; ++hi) {
    const int K = (hi < 8) ? (kg*8 + hi) : (16 + kg*8 + (hi-8));
    myK[hi] = K;
    const int Kc = (K < Tn) ? K : 0;
    const float lf = (Kc < 9) ? life0 : (Kc < 18) ? life1 : life2;
    myTao[hi] = lf + (-40.0f + 10.0f * (float)(Kc % 9));
  }

  float taoAcc[16];
  #pragma unroll
  for (int hi = 0; hi < 16; ++hi) taoAcc[hi] = 0.0f;
  float x2acc[Pn] = {0.f, 0.f, 0.f, 0.f};

  for (int p = 0; p < Pn; ++p) {
    const float Time_p = fmaxf(t[p] * ratio[p] * 1000.0f, 0.0f);

    // B-matrix (32x16 f16): lane column = m; lanes 0-15 own K {0..7, 16..23},
    // lanes 16-31 own K {8..15, 24..31}. col0 = et[k,p], col1 = valid mask, rest 0.
    v16h Bm;
    #pragma unroll
    for (int hi = 0; hi < 16; ++hi) {
      const int Kb = (hi < 8) ? (kg*8 + hi) : (16 + kg*8 + (hi-8));
      float bv = 0.0f;
      if (Kb < Tn) {
        const float lf = (Kb < 9) ? life0 : (Kb < 18) ? life1 : life2;
        const float tv = lf + (-40.0f + 10.0f * (float)(Kb % 9));
        if (m == 0)      bv = __expf(-Time_p * __builtin_amdgcn_rcpf(tv));
        else if (m == 1) bv = 1.0f;                  // sum column
      }
      Bm[hi] = (_Float16)bv;
    }

    #pragma unroll
    for (int c = 0; c < Cn; ++c) {
      // per-element randomized Life / sigma (same for both lane halves)
      const uint64_t eid = ((((uint64_t)b*Pn + p)*Cn + c) << 16) | (uint64_t)pix;
      const float u1 = u01(eid * 2 + 0x1234567ULL);
      const float u2 = u01(eid * 2 + 0x89ABCDEULL + 1);
      const float Lr = ((c == 0) ? life0 : (c == 1) ? life1 : life2)
                       + (u1 - 0.5f) * 12.0f;            // +/- 6
      const float sr = 6.0f + (u2 - 0.5f) * 0.6f;        // +/- 5% of 6
      const float rs = __builtin_amdgcn_rcpf(sr);        // v_rcp_f32 (TRANS)
      const float inv2s2 = 0.5f * rs * rs;
      const float pref = coef * rs;

      float wf[16];
      v16h Am;
      #pragma unroll
      for (int hi = 0; hi < 16; ++hi) {
        const float d = myTao[hi] - Lr;
        const float w = (myK[hi] < Tn) ? (pref * __expf(-d * d * inv2s2)) : 0.0f;
        wf[hi] = w;
        Am[hi] = (_Float16)w;
      }

      v8f cz = {0.f,0.f,0.f,0.f,0.f,0.f,0.f,0.f};
      v8f d = __builtin_amdgcn_wmma_f32_16x16x32_f16(
                  false, Am, false, Bm, (short)0, cz, false, false);

      // D layout: lane n (cols) rows 0-7 in v0-7; lanes 16+n rows 8-15
      if (lane == 0 || lane == 16) {
        const int rb = (lane == 16) ? 8 : 0;
        #pragma unroll
        for (int r = 0; r < 8; ++r) ldsNum[wave][rb + r] = d[r];
      }
      if (lane == 1 || lane == 17) {
        const int rb = (lane == 17) ? 8 : 0;
        #pragma unroll
        for (int r = 0; r < 8; ++r) ldsS[wave][rb + r] = d[r];
      }
      __syncthreads();
      const float S    = ldsS[wave][m];
      const float num  = ldsNum[wave][m];
      const float invS = __builtin_amdgcn_rcpf(S);  // S > 0 (Gaussian weights)

      #pragma unroll
      for (int hi = 0; hi < 16; ++hi) taoAcc[hi] += wf[hi] * invS;

      if (kg == 0) {
        const float xv = xTile[p*Cn + c][pixLocal];
        const float ic = (c == 0) ? int0 : (c == 1) ? int1 : int2;
        x2acc[p] += xv * (num * invS) * ic;
      }
      __syncthreads();  // LDS reused next (p,c) iteration
    }
  }

  // tao_gt[b, K, h, w] = sum_c mean_p w_norm  (x 0.25 for mean over P=4)
  #pragma unroll
  for (int hi = 0; hi < 16; ++hi) {
    if (myK[hi] < Tn) {
      tao_gt[(((size_t)b*Tn + myK[hi]) << 16) + (size_t)pix] = taoAcc[hi] * 0.25f;
    }
  }

  // x2 store + block max
  float mymax = 0.0f;
  if (kg == 0) {
    #pragma unroll
    for (int p = 0; p < Pn; ++p) {
      x2buf[(((size_t)b*Pn + p) << 16) + (size_t)pix] = x2acc[p];
      mymax = fmaxf(mymax, x2acc[p]);
    }
  }
  #pragma unroll
  for (int off = 16; off >= 1; off >>= 1)
    mymax = fmaxf(mymax, __shfl_xor(mymax, off, 32));
  if (lane == 0) ldsWMax[wave] = mymax;
  __syncthreads();
  if (threadIdx.x == 0) {
    float r = ldsWMax[0];
    #pragma unroll
    for (int wv = 1; wv < 8; ++wv) r = fmaxf(r, ldsWMax[wv]);
    blockMax[blockIdx.y * gridDim.x + blockIdx.x] = r;
  }
}

// ---------------- Pass 1b: reduce 1024 block maxima -> gmax ----------------
__global__ __launch_bounds__(256)
void decay_reduce_max(const float* __restrict__ blockMax, int n,
                      float* __restrict__ gmax)
{
  __shared__ float sm[256];
  float r = 0.0f;
  for (int i = threadIdx.x; i < n; i += 256) r = fmaxf(r, blockMax[i]);
  sm[threadIdx.x] = r;
  __syncthreads();
  for (int s = 128; s > 0; s >>= 1) {
    if ((int)threadIdx.x < s) sm[threadIdx.x] = fmaxf(sm[threadIdx.x], sm[threadIdx.x + s]);
    __syncthreads();
  }
  if (threadIdx.x == 0) gmax[0] = sm[0];
}

// ---------------- Pass 2: noise + sliced bg patch + clip ----------------
__global__ __launch_bounds__(256)
void decay_pass2(const float* __restrict__ x2buf, const float* __restrict__ gmax,
                 const float* __restrict__ bg_file, const float* __restrict__ t,
                 const float* __restrict__ ratio,
                 float* __restrict__ out_x4, float* __restrict__ out_misc)
{
  const int gid = blockIdx.x * 256 + threadIdx.x;   // b,p,i,j flattened
  const int j = gid & 255;
  const int i = (gid >> 8) & 255;
  const int p = (gid >> 16) & 3;
  const int b = gid >> 18;

  const int idx = (int)(hash32(0x2ACULL) % 9u);     // dynamic-slice row offset

  const float x2 = x2buf[gid];
  const float xm = x2 * __builtin_amdgcn_rcpf(gmax[0] + 1e-4f);

  const uint64_t base = 0x9E3779B97F4A7C15ULL ^ ((uint64_t)gid << 3);
  // Gaussian read noise (Box-Muller)
  const float un1 = fmaxf(u01(base + 1), 1e-12f);
  const float un2 = u01(base + 2);
  const float nrm = sqrtf(-2.0f * __logf(un1)) * __cosf(6.2831853f * un2);
  const float n_int = xm * 5.0f + 1.0f * nrm;                     // NOISE_MU/SIGMA
  const float p_int = poisson_knuth(fmaxf(xm, 0.0f), base + 0x100);
  const float pois  = poisson_knuth(2.0f, base + 0x200);          // NOISE_POISSON
  const float x3 = n_int + p_int + pois + x2;

  // background patch: generate bgn only inside the 256x256 slice
  const uint64_t bpk = ((uint64_t)(b * 4 + p)) * 0xD1B54A32D192ED03ULL;
  const float bg1 = u01(bpk ^ 0x1111ULL) * 0.5f + 0.5f;
  const float bg2 = u01(bpk ^ 0x2222ULL) * 0.5f + 0.5f;
  const size_t bgoff = ((size_t)p * BGHn + (size_t)(idx + i)) * BGWn + (size_t)j;
  __builtin_prefetch(&bg_file[bgoff], 0, 0);
  const float bgv = bg_file[bgoff];
  const float un3 = fmaxf(u01(base + 5), 1e-12f);
  const float un4 = u01(base + 6);
  const float bn  = sqrtf(-2.0f * __logf(un3)) * __cosf(6.2831853f * un4);
  const float patch = (bgv * bg1 + bn * bg2) * 0.1f;              // BG_RATIO

  out_x4[gid] = fminf(fmaxf(x3 + patch, 0.0f), 255.0f) * (1.0f/255.0f);

  if (gid < 4) {
    out_misc[gid]     = t[gid] * ratio[gid];   // output 3: t*ratio
    out_misc[4 + gid] = ratio[gid];            // output 4: ratio
  }
}

extern "C" void kernel_launch(void* const* d_in, const int* in_sizes, int n_in,
                              void* d_out, int out_size, void* d_ws, size_t ws_size,
                              hipStream_t stream) {
  (void)in_sizes; (void)n_in; (void)out_size; (void)ws_size;
  const float* x     = (const float*)d_in[0];
  const float* t     = (const float*)d_in[1];
  const float* ratio = (const float*)d_in[2];
  const float* life  = (const float*)d_in[3];
  const float* inten = (const float*)d_in[4];
  const float* bg    = (const float*)d_in[5];

  float* out = (float*)d_out;
  float* out_x4  = out;                                     // 524288
  float* out_tao = out + (size_t)Bn*Pn*HW;                  // 3538944
  float* out_misc = out + (size_t)Bn*Pn*HW + (size_t)Bn*Tn*HW;  // 8

  float* x2buf    = (float*)d_ws;                           // 524288 floats
  float* blockMax = x2buf + (size_t)Bn*Pn*HW;               // 1024 floats
  float* gmaxp    = blockMax + 1024;                        // 1 float

  dim3 g1(512, 2);
  decay_pass1<<<g1, 256, 0, stream>>>(x, t, ratio, life, inten,
                                      out_tao, x2buf, blockMax);
  decay_reduce_max<<<1, 256, 0, stream>>>(blockMax, 1024, gmaxp);
  decay_pass2<<<2048, 256, 0, stream>>>(x2buf, gmaxp, bg, t, ratio,
                                        out_x4, out_misc);
}
